// YOLOXHead_83648783057405
// MI455X (gfx1250) — matrix operand order repaired and checked by reading, not compiled, not run on hardware
//
#include <hip/hip_runtime.h>
#include <hip/hip_bf16.h>

typedef __attribute__((ext_vector_type(16))) _Float16 v16h;
typedef __attribute__((ext_vector_type(8)))  _Float16 v8h;
typedef __attribute__((ext_vector_type(8)))  float    v8f;
typedef __attribute__((ext_vector_type(4)))  unsigned int v4u;
typedef __attribute__((ext_vector_type(8)))  int v8i;
typedef __attribute__((ext_vector_type(4)))  int v4i;

#define NCLS   80
#define WIDTH  128
#define BB     16
#define NGTS   50
#define ATOT   8400
#define BN_EPS 1e-5f
#define LDPAD  8

__device__ __forceinline__ float sigmoidf_(float x){ return 1.0f/(1.0f+expf(-x)); }
__device__ __forceinline__ float bce_logits_(float x, float t){
  return fmaxf(x,0.0f) - x*t + log1pf(expf(-fabsf(x)));
}

// ---- WMMA fragment loaders (CDNA5 16-bit layouts, wave32) ----
// A 16x32: lane L holds row m=L&15; K = e + hi*8 (e<8) / e+8+hi*8 (e>=8), hi=L>>4
__device__ __forceinline__ v16h load_a16(const _Float16* Wrow, int kb, int hi){
  v8h lo = *(const v8h*)(Wrow + kb + hi*8);
  v8h hh = *(const v8h*)(Wrow + kb + hi*8 + 16);
  v16h a;
#pragma unroll
  for (int e=0;e<8;++e){ a[e]=lo[e]; a[e+8]=hh[e]; }
  return a;
}
// B 32x16: lane L holds col n=L&15; K = e + hi*8 (e<8) / e+8+hi*8 (e>=8), same as A
__device__ __forceinline__ v16h load_b16(const _Float16* Xrow, int kb, int hi){
  v8h lo = *(const v8h*)(Xrow + kb + hi*8);
  v8h hh = *(const v8h*)(Xrow + kb + hi*8 + 16);
  v16h b;
#pragma unroll
  for (int e=0;e<8;++e){ b[e]=lo[e]; b[e+8]=hh[e]; }
  return b;
}

// ---------------- pass 1: conv1x1 (WMMA) + bias + BN stats ----------------
// x tile (C rows of 16 consecutive f32, row stride HW) staged into LDS by the
// Tensor Data Mover (one D# descriptor, TENSORcnt-tracked), then converted
// LDS->LDS into the transposed/padded f16 WMMA-B layout.
__global__ void __launch_bounds__(256) conv_wmma_kernel(
    const float* __restrict__ x, const _Float16* __restrict__ wf16,
    const float* __restrict__ bias, _Float16* __restrict__ feat,
    float* __restrict__ stats, int C, int HW)
{
  extern __shared__ char smem[];
  float*    Xs  = (float*)smem;                                   // [C][16] f32 TDM staging
  _Float16* Xt  = (_Float16*)(smem + (size_t)C*16*4);             // [16][C+LDPAD] transposed
  float*    ssum = (float*)(smem + (size_t)C*16*4 + (size_t)16*(C+LDPAD)*2);
  float*    ssq  = ssum + WIDTH;

  const int tid   = threadIdx.x;
  const int tiles = HW >> 4;
  const int b     = blockIdx.x / tiles;
  const int p0    = (blockIdx.x % tiles) << 4;
  const int lstr  = C + LDPAD;

  if (tid < WIDTH){ ssum[tid]=0.0f; ssq[tid]=0.0f; }

#if defined(__HIP_DEVICE_COMPILE__)
  if (tid < 32){                      // wave 0 issues the TDM descriptor
    const float* src = x + (size_t)b*C*HW + p0;
    unsigned long long ga = (unsigned long long)(size_t)(const void*)src;
    unsigned ldsoff = (unsigned)(size_t)(void*)Xs;
    v4u g0;
    g0[0] = 1u;                                              // count=1 (valid D#)
    g0[1] = ldsoff;                                          // lds_addr (bytes)
    g0[2] = (unsigned)ga;                                    // global_addr[31:0]
    g0[3] = (unsigned)((ga >> 32) & 0x01ffffffu) | (2u<<30); // addr[56:32], type=2
    v8i g1;
    g1[0] = (int)(2u << 16);                                 // data_size=4B, no multicast
    g1[1] = (int)(((unsigned)HW & 0xffffu) << 16);           // tensor_dim0[15:0]
    g1[2] = (int)(((unsigned)HW >> 16) | (((unsigned)C & 0xffffu) << 16)); // tdim0 hi | tensor_dim1 lo
    g1[3] = (int)(((unsigned)C >> 16) | (16u << 16));        // tdim1 hi | tile_dim0=16
    g1[4] = (int)(unsigned)C;                                // tile_dim1=C, tile_dim2=0
    g1[5] = (int)(unsigned)HW;                               // tensor_dim0_stride[31:0]
    g1[6] = 0;
    g1[7] = 0;
    v4i g2 = {0,0,0,0};
    v4i g3 = {0,0,0,0};
#if __clang_major__ >= 23
    v8i g4 = {0,0,0,0,0,0,0,0};
    __builtin_amdgcn_tensor_load_to_lds(g0, g1, g2, g3, g4, 0);
#else
    __builtin_amdgcn_tensor_load_to_lds(g0, g1, g2, g3, 0);
#endif
    __builtin_amdgcn_s_wait_tensorcnt(0);
  }
#endif
  __syncthreads();

  // LDS->LDS: f32 [C][16] -> f16 transposed [16][C+LDPAD]
  for (int idx = tid; idx < C*16; idx += 256){
    int n = idx & 15, c = idx >> 4;
    Xt[n*lstr + c] = (_Float16)Xs[c*16 + n];
  }
  __syncthreads();

  const int wv = tid >> 5, lane = tid & 31, hi = lane >> 4, n = lane & 15;
  const int mbase = wv << 4;
  const _Float16* Wrow = wf16 + (size_t)(mbase + (lane & 15))*C;
  const _Float16* Xrow = Xt + n*lstr;

  v8f acc = {};
  for (int kb = 0; kb < C; kb += 32){
    v16h a  = load_a16(Wrow, kb, hi);
    v16h bb = load_b16(Xrow, kb, hi);
    acc = __builtin_amdgcn_wmma_f32_16x16x32_f16(false, a, false, bb, (short)0, acc, false, false);
  }

#pragma unroll
  for (int r = 0; r < 8; ++r){
    int m = mbase + r + hi*8;                 // C-frag: VGPR r -> row r (+8 for hi half)
    float v = acc[r] + bias[m];
    feat[((size_t)b*WIDTH + m)*HW + p0 + n] = (_Float16)v;
    float s1 = v, s2 = v*v;
#pragma unroll
    for (int off = 8; off >= 1; off >>= 1){   // reduce across 16-lane half
      s1 += __shfl_xor(s1, off, 32);
      s2 += __shfl_xor(s2, off, 32);
    }
    if ((lane & 15) == 0){ atomicAdd(&ssum[m], s1); atomicAdd(&ssq[m], s2); }
  }
  __syncthreads();
  if (tid < WIDTH){
    atomicAdd(&stats[tid],        ssum[tid]);
    atomicAdd(&stats[WIDTH+tid],  ssq[tid]);
  }
}

__global__ void bn_finalize_kernel(const float* __restrict__ stats,
    const float* __restrict__ gamma, const float* __restrict__ beta,
    float* __restrict__ aff, float invN)
{
  int c = threadIdx.x;
  if (c < WIDTH){
    float mean = stats[c]*invN;
    float var  = stats[WIDTH+c]*invN - mean*mean;
    float a    = gamma[c]*rsqrtf(var + BN_EPS);
    aff[c]        = a;
    aff[WIDTH+c]  = beta[c] - mean*a;
  }
}

// ---------------- pass 2: BN+SiLU fused into head GEMM (WMMA), decode to d_out ----------------
__global__ void __launch_bounds__(256) pred_wmma_kernel(
    const _Float16* __restrict__ featc, const _Float16* __restrict__ featr,
    const float* __restrict__ affc, const float* __restrict__ affr,
    const _Float16* __restrict__ fw, const float* __restrict__ fb,
    float* __restrict__ outs, int HW, int Wimg, float strd, int aoff)
{
  __shared__ _Float16 Xc[16*(WIDTH+LDPAD)];
  __shared__ _Float16 Xr[16*(WIDTH+LDPAD)];
  const int tid   = threadIdx.x;
  const int tiles = HW >> 4;
  const int b     = blockIdx.x / tiles;
  const int p0    = (blockIdx.x % tiles) << 4;
  const int lstr  = WIDTH + LDPAD;

  for (int idx = tid; idx < WIDTH*16; idx += 256){
    int n = idx & 15, ch = idx >> 4;
    size_t src = ((size_t)b*WIDTH + ch)*HW + p0 + n;
    float yc = affc[ch]*(float)featc[src] + affc[WIDTH+ch];
    float yr = affr[ch]*(float)featr[src] + affr[WIDTH+ch];
    Xc[n*lstr + ch] = (_Float16)(yc * sigmoidf_(yc));
    Xr[n*lstr + ch] = (_Float16)(yr * sigmoidf_(yr));
  }
  __syncthreads();

  const int wv = tid >> 5;
  if (wv >= 6) return;                        // 6 M-tiles: 5 cls + 1 reg/obj
  const int lane = tid & 31, hi = lane >> 4, n = lane & 15;
  const bool isreg = (wv == 5);
  const int frbase = isreg ? 0 : 16 + (wv << 4);
  const _Float16* Wrow = fw + (size_t)(frbase + (lane & 15))*WIDTH;
  const _Float16* Xrow = (isreg ? Xr : Xc) + n*lstr;

  v8f acc = {};
#pragma unroll
  for (int kb = 0; kb < WIDTH; kb += 32){
    v16h a  = load_a16(Wrow, kb, hi);
    v16h bb = load_b16(Xrow, kb, hi);
    acc = __builtin_amdgcn_wmma_f32_16x16x32_f16(false, a, false, bb, (short)0, acc, false, false);
  }

  const int p = p0 + n;
  const float gx = (float)(p % Wimg), gy = (float)(p / Wimg);
  const size_t obase = ((size_t)b*ATOT + aoff + p)*85;
#pragma unroll
  for (int r = 0; r < 8; ++r){
    int fr = frbase + r + hi*8;
    float v = acc[r] + fb[fr];
    if (fr >= 16)      outs[obase + 5 + (fr-16)] = v;           // cls logits
    else if (fr == 0)  outs[obase + 0] = (v + gx)*strd;         // x
    else if (fr == 1)  outs[obase + 1] = (v + gy)*strd;         // y
    else if (fr == 2)  outs[obase + 2] = expf(v)*strd;          // w
    else if (fr == 3)  outs[obase + 3] = expf(v)*strd;          // h
    else if (fr == 4)  outs[obase + 4] = v;                     // obj logit
  }
}

// ---------------- small prep kernels ----------------
__global__ void zerof_kernel(float* p, int n){ int i=blockIdx.x*256+threadIdx.x; if(i<n) p[i]=0.f; }

__global__ void f32_to_f16_kernel(const float* __restrict__ src, _Float16* __restrict__ dst, int n){
  int i = blockIdx.x*256 + threadIdx.x;
  if (i < n) dst[i] = (_Float16)src[i];
}

__global__ void prep_fused_kernel(const float* __restrict__ clsw, const float* __restrict__ clsb,
    const float* __restrict__ regw, const float* __restrict__ regb,
    const float* __restrict__ objw, const float* __restrict__ objb,
    _Float16* __restrict__ fw, float* __restrict__ fb)
{
  int i = blockIdx.x*256 + threadIdx.x;
  if (i < 96*WIDTH){
    int fr = i / WIDTH, k = i % WIDTH;
    float v = 0.0f;
    if (fr < 4)        v = regw[fr*WIDTH + k];
    else if (fr == 4)  v = objw[k];
    else if (fr >= 16) v = clsw[(fr-16)*WIDTH + k];
    fw[i] = (_Float16)v;
  }
  if (i < 96){
    float bv = 0.0f;
    if (i < 4)        bv = regb[i];
    else if (i == 4)  bv = objb[0];
    else if (i >= 16) bv = clsb[i-16];
    fb[i] = bv;
  }
}

__global__ void init_anchors_kernel(float* __restrict__ cen, float* __restrict__ stra){
  int a = blockIdx.x*256+threadIdx.x;
  if (a >= ATOT) return;
  int p, w; float s;
  if (a < 6400){ p=a;      w=80; s=8.f;  }
  else if (a < 8000){ p=a-6400; w=40; s=16.f; }
  else { p=a-8000; w=20; s=32.f; }
  cen[2*a]   = ((float)(p % w) + 0.5f)*s;
  cen[2*a+1] = ((float)(p / w) + 0.5f)*s;
  stra[a] = s;
}

// ---------------- SimOTA: per-anchor prep (fg mask + Sum_c log1p(-p)) ----------------
__global__ void anchor_prep_kernel(const float* __restrict__ outs, const float* __restrict__ labels,
   const float* __restrict__ cen, const float* __restrict__ stra,
   float* __restrict__ S, float* __restrict__ fg0)
{
  int idx = blockIdx.x*256+threadIdx.x;
  if (idx >= BB*ATOT) return;
  int b = idx / ATOT, a = idx % ATOT;
  const float* o = outs + (size_t)idx*85;
  float so = sigmoidf_(o[4]);
  float s = 0.f;
  for (int c=0;c<NCLS;++c){
    float p = sqrtf(sigmoidf_(o[5+c])*so);
    p = fminf(fmaxf(p, 1e-12f), 1.f-1e-12f);
    s += log1pf(-p);
  }
  S[idx] = s;
  float cx = cen[2*a], cy = cen[2*a+1], rad = 2.5f*stra[a];
  float fg = 0.f;
  for (int g=0; g<NGTS; ++g){
    const float* gt = labels + ((size_t)b*NGTS+g)*5;
    float gx=gt[0], gy=gt[1], gw=gt[2], gh=gt[3];
    bool inb = (cx > gx-0.5f*gw) && (cy > gy-0.5f*gh) && (gx+0.5f*gw > cx) && (gy+0.5f*gh > cy);
    bool inc = (fabsf(cx-gx) < rad) && (fabsf(cy-gy) < rad);
    if (inb || inc){ fg = 1.f; break; }
  }
  fg0[idx] = fg;
}

// ---------------- SimOTA: iou + cost matrices (one-hot factorized cls cost) ----------------
__global__ void cost_kernel(const float* __restrict__ outs, const float* __restrict__ labels,
   const float* __restrict__ cen, const float* __restrict__ stra,
   const float* __restrict__ S, const float* __restrict__ fg0,
   float* __restrict__ ious, float* __restrict__ cost, unsigned char* __restrict__ match)
{
  size_t idx = (size_t)blockIdx.x*256+threadIdx.x;
  if (idx >= (size_t)BB*NGTS*ATOT) return;
  int a = (int)(idx % ATOT);
  int g = (int)((idx / ATOT) % NGTS);
  int b = (int)(idx / ((size_t)ATOT*NGTS));
  const float* gt = labels + ((size_t)b*NGTS+g)*5;
  const float* o  = outs + ((size_t)b*ATOT + a)*85;
  float gx=gt[0], gy=gt[1], gw=gt[2], gh=gt[3];
  float px=o[0],  py=o[1],  pw=o[2],  ph=o[3];
  float tlx=fmaxf(gx-0.5f*gw, px-0.5f*pw), tly=fmaxf(gy-0.5f*gh, py-0.5f*ph);
  float brx=fminf(gx+0.5f*gw, px+0.5f*pw), bry=fminf(gy+0.5f*gh, py+0.5f*ph);
  float en = (tlx<brx && tly<bry) ? 1.f : 0.f;
  float ai = (brx-tlx)*(bry-tly)*en;
  float fgv = fg0[(size_t)b*ATOT+a];
  float iou = ai/(gw*gh + pw*ph - ai) * fgv;
  ious[idx] = iou;
  float iou_cost = -logf(iou + 1e-8f);
  float cx = cen[2*a], cy = cen[2*a+1], rad = 2.5f*stra[a];
  bool inb = (cx > gx-0.5f*gw) && (cy > gy-0.5f*gh) && (gx+0.5f*gw > cx) && (gy+0.5f*gh > cy);
  bool inc = (fabsf(cx-gx) < rad) && (fabsf(cy-gy) < rad);
  int gc = (int)gt[4];
  float p = sqrtf(sigmoidf_(o[5+gc])*sigmoidf_(o[4]));
  p = fminf(fmaxf(p, 1e-12f), 1.f-1e-12f);
  float cls_cost = -(logf(p) - log1pf(-p) + S[(size_t)b*ATOT+a]);
  float c = cls_cost + 3.f*iou_cost;
  if (!(inb && inc)) c += 100000.f;
  if (fgv == 0.f)    c += 100000.f;
  cost[idx]  = c;
  match[idx] = 0;
}

// ---------------- SimOTA: dyn_k + matching (one wave32 per (b,g)) ----------------
__global__ void match_kernel(const float* __restrict__ ious, const float* __restrict__ cost,
                             unsigned char* __restrict__ match)
{
  int bg   = blockIdx.x;
  int lane = threadIdx.x;
  const float* irow = ious + (size_t)bg*ATOT;
  const float* crow = cost + (size_t)bg*ATOT;
  int sel[10];
  // dyn_k = clamp(int(sum of top-10 ious), 1, 10)
  float sum10 = 0.f; int nsel = 0;
  for (int it=0; it<10; ++it){
    float bv = -1.f; int bi = 0x7fffffff;
    for (int a=lane; a<ATOT; a+=32){
      bool used=false;
      for (int j=0;j<nsel;++j) if (sel[j]==a) used=true;
      if (used) continue;
      float v = irow[a];
      if (v > bv || (v == bv && a < bi)){ bv=v; bi=a; }
    }
    for (int off=16; off>=1; off>>=1){
      float ov = __shfl_xor(bv, off, 32);
      int   oi = __shfl_xor(bi, off, 32);
      if (ov > bv || (ov == bv && oi < bi)){ bv=ov; bi=oi; }
    }
    sel[nsel++] = bi;
    sum10 += bv;
  }
  int dynk = (int)sum10; if (dynk < 1) dynk = 1; if (dynk > 10) dynk = 10;
  // dyn_k lexicographically smallest (cost, idx) -> matched
  nsel = 0;
  for (int it=0; it<dynk; ++it){
    float bv = 3.4e38f; int bi = 0x7fffffff;
    for (int a=lane; a<ATOT; a+=32){
      bool used=false;
      for (int j=0;j<nsel;++j) if (sel[j]==a) used=true;
      if (used) continue;
      float v = crow[a];
      if (v < bv || (v == bv && a < bi)){ bv=v; bi=a; }
    }
    for (int off=16; off>=1; off>>=1){
      float ov = __shfl_xor(bv, off, 32);
      int   oi = __shfl_xor(bi, off, 32);
      if (ov < bv || (ov == bv && oi < bi)){ bv=ov; bi=oi; }
    }
    sel[nsel++] = bi;
    if (lane == 0) match[(size_t)bg*ATOT + bi] = 1;
  }
}

// ---------------- SimOTA: resolve multi-assignment per anchor ----------------
__global__ void resolve_kernel(const float* __restrict__ ious, const float* __restrict__ cost,
   const unsigned char* __restrict__ match,
   float* __restrict__ fgf, int* __restrict__ mgt, float* __restrict__ piou)
{
  int idx = blockIdx.x*256+threadIdx.x;
  if (idx >= BB*ATOT) return;
  int b = idx / ATOT, a = idx % ATOT;
  int amg = 0, first = -1, bestg = 0;
  float bestc = 3.4e38f;
  for (int g=0; g<NGTS; ++g){
    size_t r = ((size_t)b*NGTS+g)*ATOT + a;
    if (match[r]){ amg++; if (first<0) first=g; }
    float c = cost[r];
    if (c < bestc){ bestc=c; bestg=g; }
  }
  int mg; float fg, pi;
  if (amg > 1){ mg=bestg; fg=1.f; pi=ious[((size_t)b*NGTS+bestg)*ATOT + a]; }
  else if (amg == 1){ mg=first; fg=1.f; pi=ious[((size_t)b*NGTS+first)*ATOT + a]; }
  else { mg=0; fg=0.f; pi=0.f; }
  fgf[idx]=fg; mgt[idx]=mg; piou[idx]=pi;
}

// ---------------- losses ----------------
__global__ void loss_kernel(const float* __restrict__ outs, const float* __restrict__ labels,
  const float* __restrict__ fgf, const int* __restrict__ mgt, const float* __restrict__ piou,
  float* __restrict__ acc)
{
  int idx = blockIdx.x*256+threadIdx.x;
  float fg=0.f, liou=0.f, lobj=0.f, lcls=0.f;
  if (idx < BB*ATOT){
    int b = idx / ATOT;
    const float* o = outs + (size_t)idx*85;
    fg = fgf[idx];
    int mg = mgt[idx];
    const float* gt = labels + ((size_t)b*NGTS+mg)*5;
    float px=o[0],py=o[1],pw=o[2],ph=o[3];
    float gx=gt[0],gy=gt[1],gw=gt[2],gh=gt[3];
    float tlx=fmaxf(px-0.5f*pw, gx-0.5f*gw), tly=fmaxf(py-0.5f*ph, gy-0.5f*gh);
    float brx=fminf(px+0.5f*pw, gx+0.5f*gw), bry=fminf(py+0.5f*ph, gy+0.5f*gh);
    float en=(tlx<brx && tly<bry)?1.f:0.f;
    float ai=(brx-tlx)*(bry-tly)*en;
    float au=pw*ph+gw*gh-ai;
    float iou=ai/(au+1e-16f);
    float ctlx=fminf(px-0.5f*pw, gx-0.5f*gw), ctly=fminf(py-0.5f*ph, gy-0.5f*gh);
    float cbrx=fmaxf(px+0.5f*pw, gx+0.5f*gw), cbry=fmaxf(py+0.5f*ph, gy+0.5f*gh);
    float ac=(cbrx-ctlx)*(cbry-ctly);
    float giou = iou - (ac-au)/fmaxf(ac,1e-16f);
    giou = fminf(fmaxf(giou,-1.f),1.f);
    liou = (1.f - giou)*fg;
    lobj = bce_logits_(o[4], fg);
    int mcls = (int)gt[4];
    float pi = piou[idx];
    float cl = 0.f;
    for (int c=0;c<NCLS;++c){
      float t = (c==mcls) ? pi : 0.f;
      cl += bce_logits_(o[5+c], t);
    }
    lcls = cl*fg;
  }
  for (int off=16; off>=1; off>>=1){
    fg   += __shfl_xor(fg,   off, 32);
    liou += __shfl_xor(liou, off, 32);
    lobj += __shfl_xor(lobj, off, 32);
    lcls += __shfl_xor(lcls, off, 32);
  }
  if ((threadIdx.x & 31) == 0){
    atomicAdd(&acc[0], fg);
    atomicAdd(&acc[1], liou);
    atomicAdd(&acc[2], lobj);
    atomicAdd(&acc[3], lcls);
  }
}

__global__ void final_loss_kernel(const float* __restrict__ acc, float* __restrict__ out0){
  float nf = fmaxf(acc[0], 1.0f);
  out0[0] = (5.f*acc[1] + acc[2] + acc[3]) / nf;
}

// ---------------- host launch ----------------
extern "C" void kernel_launch(void* const* d_in, const int* in_sizes, int n_in,
                              void* d_out, int out_size, void* d_ws, size_t ws_size,
                              hipStream_t stream)
{
  (void)in_sizes; (void)n_in; (void)out_size; (void)ws_size;
  const int   CIN[3]  = {128,256,512};
  const int   HW[3]   = {6400,1600,400};
  const int   AOFF[3] = {0,6400,8000};
  const int   WD[3]   = {80,40,20};
  const float STRF[3] = {8.f,16.f,32.f};

  const float* x[3]   = {(const float*)d_in[0], (const float*)d_in[1], (const float*)d_in[2]};
  const float* labels = (const float*)d_in[3];
  auto P = [&](int s, int j)->const float* { return (const float*)d_in[4 + s*14 + j]; };
  // j: 0 cls_conv_w 1 cls_conv_b 2 cls_bn_g 3 cls_bn_b 4 reg_conv_w 5 reg_conv_b 6 reg_bn_g 7 reg_bn_b
  //    8 cls_pred_w 9 cls_pred_b 10 reg_pred_w 11 reg_pred_b 12 obj_pred_w 13 obj_pred_b

  char* ws = (char*)d_ws;
  size_t off = 0;
  auto alloc = [&](size_t bytes)->char* {
    off = (off + 255) & ~(size_t)255;
    char* p = ws + off;
    off += bytes;
    return p;
  };

  _Float16* wcls[3]; _Float16* wreg[3]; _Float16* fw[3]; float* fb[3];
  for (int s=0;s<3;++s){
    wcls[s] = (_Float16*)alloc((size_t)WIDTH*CIN[s]*2);
    wreg[s] = (_Float16*)alloc((size_t)WIDTH*CIN[s]*2);
    fw[s]   = (_Float16*)alloc((size_t)96*WIDTH*2);
    fb[s]   = (float*)alloc(96*4);
  }
  float* stats = (float*)alloc((size_t)3*2*2*WIDTH*4);
  float* aff   = (float*)alloc((size_t)3*2*2*WIDTH*4);
  _Float16* feat[3][2];
  for (int s=0;s<3;++s) for (int br=0;br<2;++br)
    feat[s][br] = (_Float16*)alloc((size_t)BB*WIDTH*HW[s]*2);
  float* cen  = (float*)alloc((size_t)ATOT*2*4);
  float* stra = (float*)alloc((size_t)ATOT*4);
  float* Sarr = (float*)alloc((size_t)BB*ATOT*4);
  float* fg0  = (float*)alloc((size_t)BB*ATOT*4);
  float* ious = (float*)alloc((size_t)BB*NGTS*ATOT*4);
  float* cost = (float*)alloc((size_t)BB*NGTS*ATOT*4);
  unsigned char* match = (unsigned char*)alloc((size_t)BB*NGTS*ATOT);
  float* fgf  = (float*)alloc((size_t)BB*ATOT*4);
  int*   mgt  = (int*)  alloc((size_t)BB*ATOT*4);
  float* piou = (float*)alloc((size_t)BB*ATOT*4);
  float* acc  = (float*)alloc(4*4);

  float* outs = (float*)d_out + 1;   // d_out[0] = loss scalar, then outputs (B,8400,85)

  zerof_kernel<<<(3*2*2*WIDTH+255)/256, 256, 0, stream>>>(stats, 3*2*2*WIDTH);
  zerof_kernel<<<1, 256, 0, stream>>>(acc, 4);
  init_anchors_kernel<<<(ATOT+255)/256, 256, 0, stream>>>(cen, stra);

  for (int s=0;s<3;++s){
    int n = WIDTH*CIN[s];
    f32_to_f16_kernel<<<(n+255)/256,256,0,stream>>>(P(s,0), wcls[s], n);
    f32_to_f16_kernel<<<(n+255)/256,256,0,stream>>>(P(s,4), wreg[s], n);
    prep_fused_kernel<<<(96*WIDTH+255)/256,256,0,stream>>>(P(s,8),P(s,9),P(s,10),P(s,11),P(s,12),P(s,13), fw[s], fb[s]);
  }

  for (int s=0;s<3;++s){
    size_t shmem = (size_t)CIN[s]*16*4 + (size_t)16*(CIN[s]+LDPAD)*2 + (size_t)2*WIDTH*4;
    int grid = BB * (HW[s]/16);
    conv_wmma_kernel<<<grid,256,shmem,stream>>>(x[s], wcls[s], P(s,1), feat[s][0],
                                                stats + (s*2+0)*2*WIDTH, CIN[s], HW[s]);
    conv_wmma_kernel<<<grid,256,shmem,stream>>>(x[s], wreg[s], P(s,5), feat[s][1],
                                                stats + (s*2+1)*2*WIDTH, CIN[s], HW[s]);
  }

  for (int s=0;s<3;++s){
    float invN = 1.0f/((float)BB*HW[s]);
    bn_finalize_kernel<<<1,WIDTH,0,stream>>>(stats+(s*2+0)*2*WIDTH, P(s,2), P(s,3), aff+(s*2+0)*2*WIDTH, invN);
    bn_finalize_kernel<<<1,WIDTH,0,stream>>>(stats+(s*2+1)*2*WIDTH, P(s,6), P(s,7), aff+(s*2+1)*2*WIDTH, invN);
  }

  for (int s=0;s<3;++s){
    int grid = BB * (HW[s]/16);
    pred_wmma_kernel<<<grid,256,0,stream>>>(feat[s][0], feat[s][1],
                                            aff+(s*2+0)*2*WIDTH, aff+(s*2+1)*2*WIDTH,
                                            fw[s], fb[s], outs, HW[s], WD[s], STRF[s], AOFF[s]);
  }

  anchor_prep_kernel<<<(BB*ATOT+255)/256,256,0,stream>>>(outs, labels, cen, stra, Sarr, fg0);
  {
    size_t total = (size_t)BB*NGTS*ATOT;
    cost_kernel<<<(int)((total+255)/256),256,0,stream>>>(outs, labels, cen, stra, Sarr, fg0, ious, cost, match);
  }
  match_kernel<<<BB*NGTS,32,0,stream>>>(ious, cost, match);
  resolve_kernel<<<(BB*ATOT+255)/256,256,0,stream>>>(ious, cost, match, fgf, mgt, piou);
  loss_kernel<<<(BB*ATOT+255)/256,256,0,stream>>>(outs, labels, fgf, mgt, piou, acc);
  final_loss_kernel<<<1,1,0,stream>>>(acc, (float*)d_out);
}
